// ConvPoint_32847909880420
// MI455X (gfx1250) — compile-verified, matrix operand order checked by reading, not run on hardware
//
#include <hip/hip_runtime.h>
#include <hip/hip_bf16.h>
#include <cstdint>
#include <cstddef>

#define USE_ASYNC_W 1

// ---------------------------------------------------------------------------
// Types for CDNA5 WMMA (wave32): A/B are 16 f16 per lane, C/D 8 f32 per lane.
// ---------------------------------------------------------------------------
typedef _Float16 half_t;
typedef __attribute__((ext_vector_type(16))) _Float16 v16h;
typedef __attribute__((ext_vector_type(8)))  _Float16 v8h;
typedef __attribute__((ext_vector_type(8)))  float    v8f;

static __device__ __forceinline__ v8f wmma_f16(v16h a, v16h b, v8f c) {
  // (neg_a, A, neg_b, B, c_mod, C, reuse_a, reuse_b)
  return __builtin_amdgcn_wmma_f32_16x16x32_f16(false, a, false, b, (short)0, c,
                                                false, false);
}

// Element e (0..15) of a lane's 16-bit A/B fragment maps to k index:
// e in [0,7]  -> k = e + 8*hf        (contiguous)
// e in [8,15] -> k = (e-8) + 16+8*hf (contiguous)
static __host__ __device__ __forceinline__ int kmap(int e, int hf) {
  int r = e >> 1;
  int base = (r < 4) ? (r << 1) : (16 + ((r - 4) << 1));
  return base + (hf << 3) + (e & 1);
}

// Build a v16h A/B fragment from two contiguous 8-half chunks at k=8*hf and
// k=16+8*hf of a k-contiguous (32-slot, zero-padded) row.
static __device__ __forceinline__ v16h frag_from_row(const half_t* row, int hf) {
  v8h lo = *(const v8h*)(row + 8 * hf);
  v8h hi = *(const v8h*)(row + 16 + 8 * hf);
  return __builtin_shufflevector(lo, hi, 0, 1, 2, 3, 4, 5, 6, 7, 8, 9, 10, 11,
                                 12, 13, 14, 15);
}

// ---------------------------------------------------------------------------
// Fused ConvPoint layer:
//   feat  : (B, NIN, C)   f16 normalized input features
//   posIn : (B, NIN, 3)   f32 positions of input cloud
//   supp  : (B, M, 3)     f32 positions of query points
//   ids   : (B, M, K)     neighbor indices
//   Whs   : swizzled f16 W, chunk (t*NT+nt): 32 lanes x 16 halfs (fragment order)
//   out   : (B, M, COUT)  f32 raw (pre-BN) output
// Block = 256 threads (8 wave32), 16 query points per block.
// ---------------------------------------------------------------------------
template <int C, int COUT, int K>
__global__ void __launch_bounds__(256)
pconv_kernel(const half_t* __restrict__ feat, const float* __restrict__ posIn,
             const float* __restrict__ supp, const int* __restrict__ ids,
             const float* __restrict__ A1, const float* __restrict__ b1,
             const float* __restrict__ A2, const float* __restrict__ b2,
             const float* __restrict__ A3, const float* __restrict__ b3,
             const half_t* __restrict__ Whs, float* __restrict__ out,
             int M, int NIN) {
  constexpr int KDIM = ((16 * C + 31) / 32) * 32;  // padded k-extent of agg
  constexpr int NT   = COUT / 16;
  constexpr int T    = KDIM / 32;
  extern __shared__ char smem[];
  half_t* s_agg = (half_t*)smem;                         // 16*KDIM (swizzled)
  half_t* s_w16 = (half_t*)(smem + 32 * KDIM);           // [16p][16q][32k] f16
  float*  s_par = (float*)(s_w16 + 8192);                // 928
  float*  s_sup = s_par + 928;                           // 48
  int*    s_id  = (int*)(s_sup + 48);                    // 16*16 (zero-filled)
  half_t* s_stg = (half_t*)(s_id + 256);                 // 8 waves * 2 * 512

  const int tid = threadIdx.x;
  const int b   = blockIdx.y;
  const int p0  = blockIdx.x * 16;

  for (int i = tid; i < 16 * KDIM; i += 256) s_agg[i] = (half_t)0.f;
  {  // zero-fill the k-padded w table (as dwords)
    int* z = (int*)s_w16;
    for (int i = tid; i < 4096; i += 256) z[i] = 0;
  }
  for (int i = tid; i < 96;  i += 256) s_par[i]       = A1[i];
  for (int i = tid; i < 32;  i += 256) s_par[96 + i]  = b1[i];
  for (int i = tid; i < 512; i += 256) s_par[128 + i] = A2[i];
  for (int i = tid; i < 16;  i += 256) s_par[640 + i] = b2[i];
  for (int i = tid; i < 256; i += 256) s_par[656 + i] = A3[i];
  for (int i = tid; i < 16;  i += 256) s_par[912 + i] = b3[i];
  if (tid < 48) {
    int p = tid / 3, i = tid % 3;
    int m = p0 + p; if (m >= M) m = M - 1;
    s_sup[tid] = supp[((size_t)b * M + m) * 3 + i];
  }
  {  // single-writer id table: slots with k >= K hold 0 (a valid row index)
    int p = tid >> 4, k = tid & 15;
    int m = p0 + p; if (m >= M) m = M - 1;
    s_id[tid] = (k < K) ? ids[((size_t)b * M + m) * K + k] : 0;
  }
  __syncthreads();

  // ---- Phase 1: per-(point,neighbor) spatial MLP 3 -> 32 -> 16 -> 16 ----
  if (tid < 16 * K) {
    int p = tid / K, k = tid % K;
    int id = s_id[p * 16 + k];
    // warm the cache for the phase-2 feature gather of this row
    __builtin_prefetch(&feat[((size_t)b * NIN + id) * C], 0, 1);
    const float* pp = &posIn[((size_t)b * NIN + id) * 3];
    float r0 = pp[0] - s_sup[p * 3 + 0];
    float r1 = pp[1] - s_sup[p * 3 + 1];
    float r2 = pp[2] - s_sup[p * 3 + 2];
    float h1[32];
#pragma unroll
    for (int j = 0; j < 32; ++j) {
      float v = s_par[96 + j] + r0 * s_par[j] + r1 * s_par[32 + j] +
                r2 * s_par[64 + j];
      h1[j] = fmaxf(v, 0.f);
    }
    float h2[16];
#pragma unroll
    for (int q = 0; q < 16; ++q) {
      float v = s_par[640 + q];
#pragma unroll
      for (int j = 0; j < 32; ++j) v += h1[j] * s_par[128 + j * 16 + q];
      h2[q] = fmaxf(v, 0.f);
    }
#pragma unroll
    for (int q = 0; q < 16; ++q) {
      float v = s_par[912 + q];
#pragma unroll
      for (int j = 0; j < 16; ++j) v += h2[j] * s_par[656 + j * 16 + q];
      s_w16[(p * 16 + q) * 32 + k] = (half_t)v;  // w^T, k-contiguous
    }
  }
  __syncthreads();

  const int wave = tid >> 5;
  const int lane = tid & 31;
  const int ln16 = lane & 15;
  const int hf   = lane >> 4;

  // ---- Phase 2: agg(16q x C) = w^T(16 x K) @ nf(K x C) / K, via WMMA.
  // Result stored into s_agg in phase-3 A-fragment order.
  for (int p = wave; p < 16; p += 8) {
    v16h a = frag_from_row(&s_w16[(p * 16 + ln16) * 32], hf);  // zero-padded
    for (int ct = 0; ct < (C + 15) / 16; ++ct) {
      int c  = ct * 16 + ln16;
      int cc = (C % 16 == 0) ? c : (c < C ? c : C - 1);
      v16h bf;
      constexpr int LIVEE = (K == 4) ? 4 : 8;  // live fragment elements
#pragma unroll
      for (int e = 0; e < 16; ++e) {
        half_t v = (half_t)0.f;
        if (e < LIVEE) {
          int row = s_id[p * 16 + kmap(e, hf)];
          half_t lv = feat[((size_t)b * NIN + row) * C + cc];
          bool live = (K == 16) || (hf == 0);
          bool cok  = (C % 16 == 0) || (c < C);
          v = (live && cok) ? lv : (half_t)0.f;
        }
        bf[e] = v;
      }
      v8f acc;
#pragma unroll
      for (int i = 0; i < 8; ++i) acc[i] = 0.f;
      acc = wmma_f16(a, bf, acc);
      if ((C % 16 == 0) || (c < C)) {
#pragma unroll
        for (int v = 0; v < 8; ++v) {
          int q    = v + 8 * hf;
          int j    = q * C + c;            // k-position in agg_flat
          int t    = j >> 5;
          int krel = j & 31;
          int hfA  = (krel >> 3) & 1;
          int eA   = (krel & 7) | ((krel & 16) >> 1);
          int lnA  = (hfA << 4) | p;       // lane that owns A[p][j]
          s_agg[(((t << 5) | lnA) << 4) + eA] = (half_t)(acc[v] * (1.f / (float)K));
        }
      }
    }
  }
  __syncthreads();

  // ---- Phase 3: out(16m x COUT) = agg(16m x 16C) @ W(16C x COUT), via WMMA.
  // W is pre-swizzled; each wave's k-tile slice is a contiguous 1KB chunk.
  if (wave < NT) {
    const unsigned stg_base =
        (unsigned)((char*)s_stg - smem) + (unsigned)(wave * 2) * 1024u;
    const char* wbytes = (const char*)Whs;
#if USE_ASYNC_W
    {  // prime buffer 0 via async DMA
      const char* src = wbytes + (size_t)(0 * NT + wave) * 1024 + lane * 32;
      unsigned dst = stg_base + (unsigned)lane * 32u;
      asm volatile("global_load_async_to_lds_b128 %0, %1, off"
                   :: "v"(dst), "v"(src) : "memory");
      asm volatile("global_load_async_to_lds_b128 %0, %1, off offset:16"
                   :: "v"(dst), "v"(src) : "memory");
    }
#endif
    v8f acc;
#pragma unroll
    for (int i = 0; i < 8; ++i) acc[i] = 0.f;
    for (int t = 0; t < T; ++t) {
      v16h a = *(const v16h*)&s_agg[(((t << 5) + lane) << 4)];
      v16h bw;
#if USE_ASYNC_W
      int buf = t & 1;
      if (t + 1 < T) {
        const char* src = wbytes + (size_t)((t + 1) * NT + wave) * 1024 + lane * 32;
        unsigned dst = stg_base + (unsigned)((buf ^ 1) * 1024 + lane * 32);
        asm volatile("global_load_async_to_lds_b128 %0, %1, off"
                     :: "v"(dst), "v"(src) : "memory");
        asm volatile("global_load_async_to_lds_b128 %0, %1, off offset:16"
                     :: "v"(dst), "v"(src) : "memory");
        asm volatile("s_wait_asynccnt 0x2" ::: "memory");
      } else {
        asm volatile("s_wait_asynccnt 0x0" ::: "memory");
      }
      bw = *(const v16h*)(smem + stg_base + (unsigned)(buf * 1024 + lane * 32));
#else
      if (t + 1 < T)
        __builtin_prefetch(wbytes + (size_t)((t + 1) * NT + wave) * 1024 + lane * 32,
                           0, 1);
      bw = *(const v16h*)(wbytes + (size_t)(t * NT + wave) * 1024 + lane * 32);
#endif
      acc = wmma_f16(a, bw, acc);
    }
#pragma unroll
    for (int v = 0; v < 8; ++v) {
      int m = v + 8 * hf;
      if (p0 + m < M)
        out[((size_t)b * M + p0 + m) * COUT + wave * 16 + ln16] = acc[v];
    }
  }
}

// ---------------------------------------------------------------------------
// BatchNorm (training-mode batch stats over rows) + ReLU -> f16 features.
// ---------------------------------------------------------------------------
__global__ void zero_kernel(float* __restrict__ p, int n) {
  int i = blockIdx.x * blockDim.x + threadIdx.x;
  if (i < n) p[i] = 0.f;
}

__global__ void bn_reduce_kernel(const float* __restrict__ y,
                                 float* __restrict__ stats, int rows, int C) {
  __shared__ float s[256];
  for (int i = threadIdx.x; i < 2 * C; i += blockDim.x) s[i] = 0.f;
  __syncthreads();
  size_t total = (size_t)rows * C;
  for (size_t idx = (size_t)blockIdx.x * blockDim.x + threadIdx.x; idx < total;
       idx += (size_t)gridDim.x * blockDim.x) {
    int c = (int)(idx % C);
    float v = y[idx];
    atomicAdd(&s[c], v);
    atomicAdd(&s[C + c], v * v);
  }
  __syncthreads();
  for (int i = threadIdx.x; i < 2 * C; i += blockDim.x) atomicAdd(&stats[i], s[i]);
}

__global__ void bn_apply_kernel(const float* __restrict__ y,
                                half_t* __restrict__ o,
                                const float* __restrict__ stats,
                                const float* __restrict__ g,
                                const float* __restrict__ bb, int rows, int C) {
  size_t total = (size_t)rows * C;
  for (size_t idx = (size_t)blockIdx.x * blockDim.x + threadIdx.x; idx < total;
       idx += (size_t)gridDim.x * blockDim.x) {
    int c = (int)(idx % C);
    float inv = 1.f / (float)rows;
    float mean = stats[c] * inv;
    float var  = stats[C + c] * inv - mean * mean;
    float v = (y[idx] - mean) * rsqrtf(var + 1e-5f) * g[c] + bb[c];
    o[idx] = (half_t)fmaxf(v, 0.f);
  }
}

__global__ void transpose_b3n_kernel(const float* __restrict__ src,
                                     float* __restrict__ dst, int B, int N) {
  size_t total = (size_t)B * 3 * N;
  for (size_t idx = (size_t)blockIdx.x * blockDim.x + threadIdx.x; idx < total;
       idx += (size_t)gridDim.x * blockDim.x) {
    int b = (int)(idx / ((size_t)3 * N));
    int r = (int)(idx % ((size_t)3 * N));
    int ch = r / N, n = r % N;
    dst[((size_t)b * N + n) * 3 + ch] = src[idx];
  }
}

__global__ void transpose_b3n_f16_kernel(const float* __restrict__ src,
                                         half_t* __restrict__ dst, int B, int N) {
  size_t total = (size_t)B * 3 * N;
  for (size_t idx = (size_t)blockIdx.x * blockDim.x + threadIdx.x; idx < total;
       idx += (size_t)gridDim.x * blockDim.x) {
    int b = (int)(idx / ((size_t)3 * N));
    int r = (int)(idx % ((size_t)3 * N));
    int ch = r / N, n = r % N;
    dst[((size_t)b * N + n) * 3 + ch] = (half_t)src[idx];
  }
}

// W (16C x COUT) f32 -> swizzled fragment-order f16:
// out[((t*NT+nt)*32 + lane)*16 + e] = W[(t*32 + kmap(e,lane>>4))*COUT + nt*16 + lane&15]
__global__ void wswizzle_kernel(const float* __restrict__ W,
                                half_t* __restrict__ o, int KC, int COUT,
                                int NT, int total) {
  int idx = blockIdx.x * blockDim.x + threadIdx.x;
  if (idx >= total) return;
  int e     = idx & 15;
  int lane  = (idx >> 4) & 31;
  int chunk = idx >> 9;
  int nt    = chunk % NT;
  int t     = chunk / NT;
  int k     = t * 32 + kmap(e, lane >> 4);
  int n     = nt * 16 + (lane & 15);
  o[idx] = (k < KC) ? (half_t)W[(size_t)k * COUT + n] : (half_t)0.f;
}

// fcout.W (13,128) f32 -> B-fragment order f16 (128 x 16, cols >= 13 zero)
__global__ void fcswizzle_kernel(const float* __restrict__ Wfc,
                                 half_t* __restrict__ o) {
  int idx = blockIdx.x * blockDim.x + threadIdx.x;
  if (idx >= 2048) return;
  int e    = idx & 15;
  int lane = (idx >> 4) & 31;
  int t    = idx >> 9;
  int n    = lane & 15;                 // output channel o
  int k    = t * 32 + kmap(e, lane >> 4);  // input channel c
  o[idx] = (n < 13) ? (half_t)Wfc[n * 128 + k] : (half_t)0.f;
}

__global__ void concat_kernel(const half_t* __restrict__ a,
                              const half_t* __restrict__ c2,
                              half_t* __restrict__ dst, int rows, int Ca, int Cb) {
  int Ct = Ca + Cb;
  size_t total = (size_t)rows * Ct;
  for (size_t idx = (size_t)blockIdx.x * blockDim.x + threadIdx.x; idx < total;
       idx += (size_t)gridDim.x * blockDim.x) {
    int r = (int)(idx / Ct);
    int c = (int)(idx % Ct);
    dst[idx] = (c < Ca) ? a[(size_t)r * Ca + c] : c2[(size_t)r * Cb + (c - Ca)];
  }
}

// Final projection out(B,13,N) = feat(B*N,128) @ fc^T + bias, via WMMA.
// Each wave handles a 16-point tile; grid*8 waves == (B*N)/16 tiles exactly.
__global__ void __launch_bounds__(256)
final_wmma_kernel(const half_t* __restrict__ feat,   // (B*N, 128) f16
                  const half_t* __restrict__ Bfc,    // fragment-order (4*512)
                  const float* __restrict__ bfc,     // (13)
                  float* __restrict__ out,            // (B, 13, N)
                  int B, int N) {
  const int wave = threadIdx.x >> 5;
  const int lane = threadIdx.x & 31;
  const int ln16 = lane & 15;
  const int hf   = lane >> 4;
  const size_t pbase = ((size_t)blockIdx.x * 8 + wave) * 16;
  const half_t* frow = &feat[(pbase + ln16) * 128];
  v8f acc;
#pragma unroll
  for (int i = 0; i < 8; ++i) acc[i] = 0.f;
#pragma unroll
  for (int t = 0; t < 4; ++t) {
    v16h a  = frag_from_row(frow + t * 32, hf);
    v16h bw = *(const v16h*)&Bfc[(t * 32 + lane) * 16];
    acc = wmma_f16(a, bw, acc);
  }
  const int o = ln16;
  if (o < 13) {
    float bias = bfc[o];
#pragma unroll
    for (int v = 0; v < 8; ++v) {
      size_t pidx = pbase + v + 8 * hf;
      int b = (int)(pidx / N), n = (int)(pidx % N);
      out[((size_t)b * 13 + o) * N + n] = acc[v] + bias;
    }
  }
}

// ---------------------------------------------------------------------------
// Host orchestration
// ---------------------------------------------------------------------------
extern "C" void kernel_launch(void* const* d_in, const int* in_sizes, int n_in,
                              void* d_out, int out_size, void* d_ws,
                              size_t ws_size, hipStream_t stream) {
  (void)in_sizes; (void)out_size; (void)ws_size;
  const int B = 8, N0 = 8192;
  static const int Ms[7] = {8192, 2048, 1024, 256, 64, 16, 4};

  const float* x   = (const float*)d_in[0];
  const float* pos = (const float*)d_in[1];
  const float* sup[7];
  for (int i = 0; i < 7; ++i) sup[i] = (const float*)d_in[2 + i];
  const int* idsE[7];
  for (int i = 0; i < 7; ++i) idsE[i] = (const int*)d_in[9 + i];
  const int* idsD[6];  // ids5d, ids4d, ids3d, ids2d, ids1d, ids0d
  for (int i = 0; i < 6; ++i) idsD[i] = (const int*)d_in[16 + i];

  struct CVp { const float *A1, *b1, *A2, *b2, *A3, *b3, *W; };
  struct BNp { const float *g, *b; };
  CVp cv[13]; BNp bn[13];
  const float *fcW = nullptr, *fcb = nullptr;
  static const int ci[13] = {3, 64, 64, 64, 64, 128, 128, 128, 256, 256, 128, 128, 128};
  static const int co[13] = {64, 64, 64, 64, 128, 128, 128, 128, 128, 64, 64, 64, 64};

  if (n_in <= 23) {
    // params passed as one flat blob, insertion order of setup_inputs().
    const float* base = (const float*)d_in[22];
    size_t off = 0;
    for (int li = 0; li < 13; ++li) {
      cv[li].A1 = base + off; off += 96;
      cv[li].b1 = base + off; off += 32;
      cv[li].A2 = base + off; off += 512;
      cv[li].b2 = base + off; off += 16;
      cv[li].A3 = base + off; off += 256;
      cv[li].b3 = base + off; off += 16;
      cv[li].W  = base + off; off += (size_t)16 * ci[li] * co[li];
      bn[li].g  = base + off; off += co[li];
      bn[li].b  = base + off; off += co[li];
    }
    fcW = base + off; off += 13 * 128;
    fcb = base + off; off += 13;
  } else {
    // params pytree flattened leaf-per-input, JAX sorted-key order:
    // bn0,bn0d,...,bn6 (b,g) ; cv0,cv0d,...,cv6 (A1,A2,A3,W,b1,b2,b3) ; fcout (W,b)
    static const int ord[13] = {0, 12, 1, 11, 2, 10, 3, 9, 4, 8, 5, 7, 6};
    int idx = 22;
    for (int j = 0; j < 13; ++j) {
      int li = ord[j];
      bn[li].b = (const float*)d_in[idx++];
      bn[li].g = (const float*)d_in[idx++];
    }
    for (int j = 0; j < 13; ++j) {
      int li = ord[j];
      cv[li].A1 = (const float*)d_in[idx++];
      cv[li].A2 = (const float*)d_in[idx++];
      cv[li].A3 = (const float*)d_in[idx++];
      cv[li].W  = (const float*)d_in[idx++];
      cv[li].b1 = (const float*)d_in[idx++];
      cv[li].b2 = (const float*)d_in[idx++];
      cv[li].b3 = (const float*)d_in[idx++];
    }
    fcW = (const float*)d_in[idx++];
    fcb = (const float*)d_in[idx++];
  }

  auto kdim_of = [](int Cc) { return ((16 * Cc + 31) / 32) * 32; };

  // ---- workspace bump allocator ----
  char* wsp = (char*)d_ws;
  size_t off = 0;
  auto alloc = [&](size_t elems, size_t esz) -> void* {
    void* p = wsp + off;
    off += ((elems * esz + 255) & ~(size_t)255);
    return p;
  };

  float* posT = (float*)alloc((size_t)B * N0 * 3, 4);
  float* supT[7];
  for (int i = 0; i < 7; ++i) supT[i] = (float*)alloc((size_t)B * Ms[i] * 3, 4);
  half_t* xT = (half_t*)alloc((size_t)B * N0 * 3, 2);
  static const int ceo[7] = {64, 64, 64, 64, 128, 128, 128};
  half_t* xe[7];
  for (int i = 0; i < 7; ++i) xe[i] = (half_t*)alloc((size_t)B * Ms[i] * ceo[i], 2);
  static const int Md[6] = {16, 64, 256, 1024, 2048, 8192};
  static const int Cd[6] = {128, 128, 64, 64, 64, 64};
  half_t* yd[6];
  half_t* cat[6];
  for (int i = 0; i < 6; ++i) {
    yd[i]  = (half_t*)alloc((size_t)B * Md[i] * Cd[i], 2);
    cat[i] = (half_t*)alloc((size_t)B * Md[i] * 2 * Cd[i], 2);
  }
  float* raw   = (float*)alloc((size_t)B * 8192 * 64, 4);  // max rows*COUT
  float* stats = (float*)alloc(256, 4);
  half_t* Whs[13];
  for (int li = 0; li < 13; ++li)
    Whs[li] = (half_t*)alloc((size_t)kdim_of(ci[li]) * co[li], 2);
  half_t* Bfc = (half_t*)alloc(2048, 2);

  // ---- position/feature transposes (B,3,N) -> (B,N,3) ----
  {
    int bl = (int)(((size_t)B * 3 * N0 + 255) / 256);
    transpose_b3n_kernel<<<bl, 256, 0, stream>>>(pos, posT, B, N0);
    transpose_b3n_f16_kernel<<<bl, 256, 0, stream>>>(x, xT, B, N0);
    for (int i = 0; i < 7; ++i) {
      int bl2 = (int)(((size_t)B * 3 * Ms[i] + 255) / 256);
      transpose_b3n_kernel<<<bl2, 256, 0, stream>>>(sup[i], supT[i], B, Ms[i]);
    }
  }
  // ---- projection weights f32 -> swizzled f16 ----
  for (int li = 0; li < 13; ++li) {
    int total = kdim_of(ci[li]) * co[li];
    wswizzle_kernel<<<(total + 255) / 256, 256, 0, stream>>>(
        cv[li].W, Whs[li], 16 * ci[li], co[li], co[li] / 16, total);
  }
  fcswizzle_kernel<<<8, 256, 0, stream>>>(fcW, Bfc);

  auto run_bn = [&](const float* y, half_t* o, int li, int rows, int C) {
    zero_kernel<<<1, 256, 0, stream>>>(stats, 2 * C);
    size_t tot = (size_t)rows * C;
    int blocks = (int)((tot + 255) / 256);
    if (blocks > 1024) blocks = 1024;
    bn_reduce_kernel<<<blocks, 256, 0, stream>>>(y, stats, rows, C);
    int blocks2 = (int)((tot + 255) / 256);
    if (blocks2 > 4096) blocks2 = 4096;
    bn_apply_kernel<<<blocks2, 256, 0, stream>>>(y, o, stats, bn[li].g, bn[li].b,
                                                 rows, C);
  };

#define PCONV(Cc, CO, Kk, li, featp, posp, supb, idsp, Mq, NI, outp)           \
  do {                                                                         \
    constexpr int kd_ = ((16 * (Cc) + 31) / 32) * 32;                          \
    size_t sm_ = (size_t)32 * kd_ + 16384 + 3712 + 192 + 1024 + 16384;         \
    dim3 g_(((Mq) + 15) / 16, B);                                              \
    pconv_kernel<Cc, CO, Kk><<<g_, 256, sm_, stream>>>(                        \
        featp, posp, supb, idsp, cv[li].A1, cv[li].b1, cv[li].A2, cv[li].b2,   \
        cv[li].A3, cv[li].b3, Whs[li], raw, (Mq), (NI));                       \
    run_bn(raw, outp, li, B * (Mq), CO);                                       \
  } while (0)

  // ---- encoder ----
  PCONV(3,   64,  16, 0, xT,    posT,    supT[0], idsE[0], Ms[0], N0,    xe[0]);
  PCONV(64,  64,  16, 1, xe[0], supT[0], supT[1], idsE[1], Ms[1], Ms[0], xe[1]);
  PCONV(64,  64,  16, 2, xe[1], supT[1], supT[2], idsE[2], Ms[2], Ms[1], xe[2]);
  PCONV(64,  64,  16, 3, xe[2], supT[2], supT[3], idsE[3], Ms[3], Ms[2], xe[3]);
  PCONV(64,  128, 16, 4, xe[3], supT[3], supT[4], idsE[4], Ms[4], Ms[3], xe[4]);
  PCONV(128, 128, 16, 5, xe[4], supT[4], supT[5], idsE[5], Ms[5], Ms[4], xe[5]);
  PCONV(128, 128, 16, 6, xe[5], supT[5], supT[6], idsE[6], Ms[6], Ms[5], xe[6]);

  auto run_cat = [&](const half_t* a, const half_t* b2, half_t* dst, int rows,
                     int Ca, int Cb) {
    size_t tot = (size_t)rows * (Ca + Cb);
    int bl = (int)((tot + 255) / 256);
    if (bl > 8192) bl = 8192;
    concat_kernel<<<bl, 256, 0, stream>>>(a, b2, dst, rows, Ca, Cb);
  };

  // ---- decoder ----
  PCONV(128, 128, 4, 7, xe[6], supT[6], supT[5], idsD[0], Ms[5], Ms[6], yd[0]);
  run_cat(yd[0], xe[5], cat[0], B * Ms[5], 128, 128);
  PCONV(256, 128, 4, 8, cat[0], supT[5], supT[4], idsD[1], Ms[4], Ms[5], yd[1]);
  run_cat(yd[1], xe[4], cat[1], B * Ms[4], 128, 128);
  PCONV(256, 64, 4, 9, cat[1], supT[4], supT[3], idsD[2], Ms[3], Ms[4], yd[2]);
  run_cat(yd[2], xe[3], cat[2], B * Ms[3], 64, 64);
  PCONV(128, 64, 8, 10, cat[2], supT[3], supT[2], idsD[3], Ms[2], Ms[3], yd[3]);
  run_cat(yd[3], xe[2], cat[3], B * Ms[2], 64, 64);
  PCONV(128, 64, 8, 11, cat[3], supT[2], supT[1], idsD[4], Ms[1], Ms[2], yd[4]);
  run_cat(yd[4], xe[1], cat[4], B * Ms[1], 64, 64);
  PCONV(128, 64, 8, 12, cat[4], supT[1], supT[0], idsD[5], Ms[0], Ms[1], yd[5]);
  run_cat(yd[5], xe[0], cat[5], B * Ms[0], 64, 64);

  // ---- final 13x128 projection via WMMA ----
  final_wmma_kernel<<<(B * N0) / 128, 256, 0, stream>>>(
      cat[5], Bfc, fcb, (float*)d_out, B, N0);

#undef PCONV
}
